// LRUEmbedding_15719580303536
// MI455X (gfx1250) — matrix operand order repaired and checked by reading, not compiled, not run on hardware
//
#include <hip/hip_runtime.h>

// ---------------- problem constants ----------------
constexpr int kB   = 16;
constexpr int kT   = 4096;
constexpr int kIN  = 32;
constexpr int kH   = 256;
constexpr int kN   = 256;   // LRU state dim
constexpr int kL   = 4;
constexpr int kOUT = 128;
constexpr int kMT  = kB * kT;      // 65536 GEMM rows

// GEMM tiling: block tile 64x128, K-step 32, 8 waves (wave32), wave tile 32x32
constexpr int BM = 64;
constexpr int BN = 128;
constexpr int BK = 32;

// scan chunking
constexpr int kCH  = 128;
constexpr int kNCH = kT / kCH;     // 32

typedef __attribute__((ext_vector_type(8)))  __bf16 v8bf;
typedef __attribute__((ext_vector_type(16))) __bf16 v16bf;
typedef __attribute__((ext_vector_type(8)))  float  v8f;
typedef __attribute__((ext_vector_type(4)))  int    v4i;
typedef __attribute__((address_space(1))) v4i as1_v4i;   // global int4
typedef __attribute__((address_space(3))) v4i as3_v4i;   // LDS int4

// ---------------- CDNA5 async global->LDS (guarded) ----------------
#if defined(__has_builtin)
#  if __has_builtin(__builtin_amdgcn_global_load_async_to_lds_b128) && \
      __has_builtin(__builtin_amdgcn_s_wait_asynccnt)
#    define ASYNC_LDS 1
#  endif
#endif
#ifndef ASYNC_LDS
#  define ASYNC_LDS 0
#endif

__device__ __forceinline__ __bf16 f2bf(float f) {
  unsigned u = __builtin_bit_cast(unsigned, f);
  unsigned r = u + 0x7FFFu + ((u >> 16) & 1u);   // round-to-nearest-even
  unsigned short h = (unsigned short)(r >> 16);
  return __builtin_bit_cast(__bf16, h);
}

__device__ __forceinline__ void cp_b128_to_lds(__bf16* lds_dst, const __bf16* gsrc) {
#if ASYNC_LDS
  __builtin_amdgcn_global_load_async_to_lds_b128(
      (as1_v4i*)(gsrc),
      (as3_v4i*)(lds_dst), 0, 0);
#else
  *(v8bf*)lds_dst = *(const v8bf*)gsrc;
#endif
}

// A fragment, 16x32 bf16 (ISA 7.12.2): M = lane&15; lanes>=16 take K offset +8;
// per lane two contiguous 8-element runs (K base and K base+16).
__device__ __forceinline__ v16bf load_frag_a(const __bf16* sA, int row0) {
  int lane = threadIdx.x & 31;
  const __bf16* p = sA + (size_t)(row0 + (lane & 15)) * BK + ((lane >> 4) << 3);
  v8bf lo = *(const v8bf*)(p);
  v8bf hi = *(const v8bf*)(p + 16);
  return __builtin_shufflevector(lo, hi, 0,1,2,3,4,5,6,7,8,9,10,11,12,13,14,15);
}

// B fragment, 32x16 bf16: LDS holds B col-major [n][k]; lane = column (mod 16),
// lanes>=16 take K offset +8; two 8-element runs (K base and K base+16) -> two ds_load_b128.
__device__ __forceinline__ v16bf load_frag_b(const __bf16* sB, int col0) {
  int lane = threadIdx.x & 31;
  const __bf16* p = sB + (size_t)(col0 + (lane & 15)) * BK + ((lane >> 4) << 3);
  v8bf lo = *(const v8bf*)(p);
  v8bf hi = *(const v8bf*)(p + 16);
  return __builtin_shufflevector(lo, hi, 0,1,2,3,4,5,6,7,8,9,10,11,12,13,14,15);
}

// ---------------- generic dual-B (optionally dual-A) WMMA GEMM ----------------
// mode 0: out0 = A0@B0t^T, out1 = A0@B1t^T                    (f32 stores)
// mode 1: v = A0@B0t^T - A1@B1t^T + resid*dvec; gelu -> outbf (bf16)
// mode 2: v = (A0@B0t^T + b1)*sigmoid(A0@B1t^T + b2) + hio; hio=v; outbf=bf16(v)
__global__ __launch_bounds__(256)
void gemm_dual(const __bf16* __restrict__ A0, const __bf16* __restrict__ A1,
               const __bf16* __restrict__ Bt0, const __bf16* __restrict__ Bt1,
               int Kd, int Ncols, int mode,
               float* __restrict__ out0, float* __restrict__ out1,
               const float* __restrict__ resid, const float* __restrict__ dvec,
               __bf16* __restrict__ outbf,
               const float* __restrict__ bias1, const float* __restrict__ bias2,
               float* __restrict__ hio) {
  __shared__ __align__(16) __bf16 sA0[BM * BK];
  __shared__ __align__(16) __bf16 sA1[BM * BK];
  __shared__ __align__(16) __bf16 sB0[BN * BK];
  __shared__ __align__(16) __bf16 sB1[BN * BK];

  const int m0 = blockIdx.y * BM;
  const int n0 = blockIdx.x * BN;
  const int t  = threadIdx.x;
  const int wave = t >> 5;
  const int wm = (wave & 1) * 32;   // wave M offset inside tile
  const int wn = (wave >> 1) * 32;  // wave N offset inside tile
  const bool dualA = (A1 != nullptr);

  v8f acc0[2][2] = {};
  v8f acc1[2][2] = {};

  const int ar = t >> 2, ak = (t & 3) << 3;   // A: 64 rows x 32 k, 8 bf16/thread
  const int bn = t >> 1, bk = (t & 1) << 4;   // B: 128 n x 32 k, 16 bf16/thread

  for (int k0 = 0; k0 < Kd; k0 += BK) {
    // ---- stage tiles into LDS (async where available) ----
    const __bf16* ga0 = A0 + (size_t)(m0 + ar) * Kd + k0 + ak;
    cp_b128_to_lds(&sA0[ar * BK + ak], ga0);
    if (dualA) {
      const __bf16* ga1 = A1 + (size_t)(m0 + ar) * Kd + k0 + ak;
      cp_b128_to_lds(&sA1[ar * BK + ak], ga1);
    }
    const __bf16* gb0 = Bt0 + (size_t)(n0 + bn) * Kd + k0 + bk;
    cp_b128_to_lds(&sB0[bn * BK + bk], gb0);
    cp_b128_to_lds(&sB0[bn * BK + bk + 8], gb0 + 8);
    const __bf16* gb1 = Bt1 + (size_t)(n0 + bn) * Kd + k0 + bk;
    cp_b128_to_lds(&sB1[bn * BK + bk], gb1);
    cp_b128_to_lds(&sB1[bn * BK + bk + 8], gb1 + 8);

    if (k0 + BK < Kd)  // hint next A tile toward L2
      __builtin_prefetch(A0 + (size_t)(m0 + ar) * Kd + k0 + BK + ak, 0, 1);

#if ASYNC_LDS
    __builtin_amdgcn_s_wait_asynccnt(0);
#endif
    __syncthreads();

    // ---- fragments + WMMA ----
    v16bf a0f[2], a1f[2], b0f[2], b1f[2];
#pragma unroll
    for (int mi = 0; mi < 2; ++mi) {
      a0f[mi] = load_frag_a(sA0, wm + mi * 16);
      a1f[mi] = dualA ? load_frag_a(sA1, wm + mi * 16) : a0f[mi];
    }
#pragma unroll
    for (int ni = 0; ni < 2; ++ni) {
      b0f[ni] = load_frag_b(sB0, wn + ni * 16);
      b1f[ni] = load_frag_b(sB1, wn + ni * 16);
    }
#pragma unroll
    for (int mi = 0; mi < 2; ++mi)
#pragma unroll
      for (int ni = 0; ni < 2; ++ni) {
        acc0[mi][ni] = __builtin_amdgcn_wmma_f32_16x16x32_bf16(
            false, a0f[mi], false, b0f[ni], (short)0, acc0[mi][ni], false, false);
        acc1[mi][ni] = __builtin_amdgcn_wmma_f32_16x16x32_bf16(
            false, a1f[mi], false, b1f[ni], (short)0, acc1[mi][ni], false, false);
      }
    __syncthreads();
  }

  // ---- epilogue (C/D layout: lane&15 = column, VGPR e = row, +8 for lanes 16-31) ----
  const int lane = t & 31;
  const int col = lane & 15;
  const int rowoff = (lane >> 4) << 3;
#pragma unroll
  for (int mi = 0; mi < 2; ++mi)
#pragma unroll
    for (int ni = 0; ni < 2; ++ni)
#pragma unroll
      for (int e = 0; e < 8; ++e) {
        int gm = m0 + wm + mi * 16 + rowoff + e;
        int gn = n0 + wn + ni * 16 + col;
        size_t o = (size_t)gm * Ncols + gn;
        float p0 = acc0[mi][ni][e];
        float p1 = acc1[mi][ni][e];
        if (mode == 0) {
          out0[o] = p0;
          out1[o] = p1;
        } else if (mode == 1) {
          float v = p0 - p1 + resid[o] * dvec[gn];
          float g = 0.5f * v * (1.0f + erff(v * 0.70710678118654752f));
          outbf[o] = f2bf(g);
        } else {
          float v = (p0 + bias1[gn]) *
                        (1.0f / (1.0f + __expf(-(p1 + bias2[gn])))) + hio[o];
          hio[o] = v;
          outbf[o] = f2bf(v);
        }
      }
}

// ---------------- embedding: [MT,32] @ [32,256] + b ----------------
__global__ void embed_kernel(const float* __restrict__ x, const float* __restrict__ ew,
                             const float* __restrict__ eb,
                             float* __restrict__ h, __bf16* __restrict__ hbf) {
  int m = blockIdx.x;
  int j = threadIdx.x;
  const float* xr = x + (size_t)m * kIN;
  float acc = eb[j];
#pragma unroll
  for (int k = 0; k < kIN; ++k) acc += xr[k] * ew[k * kH + j];
  size_t o = (size_t)m * kH + j;
  h[o] = acc;
  hbf[o] = f2bf(acc);
}

// ---------------- chunked complex diagonal scan ----------------
// phase 1: local scan per (b, chunk, n); overwrites u with local states; emits carry
__global__ void scan_local_kernel(float* __restrict__ uRe, float* __restrict__ uIm,
                                  const float* __restrict__ log_nu,
                                  const float* __restrict__ log_theta,
                                  float4* __restrict__ carry) {
  int idx = blockIdx.x * blockDim.x + threadIdx.x;  // (b*kNCH+ch)*kN + n
  int n = idx % kN;
  int bc = idx / kN;
  int ch = bc % kNCH;
  int b = bc / kNCH;
  float th = __expf(log_theta[n]);
  float mag = __expf(-__expf(log_nu[n]));
  float lre = mag * __cosf(th), lim = mag * __sinf(th);
  float sre = 0.f, sim = 0.f, are = 1.f, aim = 0.f;
  size_t base = ((size_t)b * kT + (size_t)ch * kCH) * kN + n;
  for (int tt = 0; tt < kCH; ++tt) {
    size_t o = base + (size_t)tt * kN;
    float ur = uRe[o], ui = uIm[o];
    float nr = lre * sre - lim * sim + ur;
    float ni = lre * sim + lim * sre + ui;
    sre = nr; sim = ni;
    uRe[o] = sre; uIm[o] = sim;
    float ar2 = lre * are - lim * aim;
    float ai2 = lre * aim + lim * are;
    are = ar2; aim = ai2;
  }
  carry[idx] = make_float4(are, aim, sre, sim);
}

// phase 2: serial combine of 32 chunk carries -> per-chunk entering state
__global__ void scan_carry_kernel(const float4* __restrict__ carry,
                                  float2* __restrict__ pref) {
  int idx = blockIdx.x * blockDim.x + threadIdx.x;  // b*kN + n
  int n = idx % kN;
  int b = idx / kN;
  float pre = 0.f, pim = 0.f;
  for (int ch = 0; ch < kNCH; ++ch) {
    size_t o = ((size_t)b * kNCH + ch) * kN + n;
    pref[o] = make_float2(pre, pim);
    float4 c = carry[o];
    float nr = c.x * pre - c.y * pim + c.z;
    float ni = c.x * pim + c.y * pre + c.w;
    pre = nr; pim = ni;
  }
}

// phase 3: s_t = s_local_t + lam^(t+1) * prefix ; write bf16 states
__global__ void scan_apply_kernel(const float* __restrict__ uRe, const float* __restrict__ uIm,
                                  const float* __restrict__ log_nu,
                                  const float* __restrict__ log_theta,
                                  const float2* __restrict__ pref,
                                  __bf16* __restrict__ stRe, __bf16* __restrict__ stIm) {
  int idx = blockIdx.x * blockDim.x + threadIdx.x;
  int n = idx % kN;
  int bc = idx / kN;
  int ch = bc % kNCH;
  int b = bc / kNCH;
  float th = __expf(log_theta[n]);
  float mag = __expf(-__expf(log_nu[n]));
  float lre = mag * __cosf(th), lim = mag * __sinf(th);
  float2 p = pref[idx];
  float are = lre, aim = lim;  // lam^(t+1), t=0
  size_t base = ((size_t)b * kT + (size_t)ch * kCH) * kN + n;
  for (int tt = 0; tt < kCH; ++tt) {
    size_t o = base + (size_t)tt * kN;
    float sr = uRe[o] + are * p.x - aim * p.y;
    float si = uIm[o] + are * p.y + aim * p.x;
    stRe[o] = f2bf(sr);
    stIm[o] = f2bf(si);
    float ar2 = lre * are - lim * aim;
    float ai2 = lre * aim + lim * are;
    are = ar2; aim = ai2;
  }
}

// ---------------- weight prep: bf16, B operand stored [col][k] ----------------
__global__ void prep_weights_kernel(const float* __restrict__ log_gamma,
                                    const float* __restrict__ B_re, const float* __restrict__ B_im,
                                    const float* __restrict__ C_re, const float* __restrict__ C_im,
                                    const float* __restrict__ w1, const float* __restrict__ w2,
                                    __bf16* __restrict__ gBre, __bf16* __restrict__ gBim,
                                    __bf16* __restrict__ Ctre, __bf16* __restrict__ Ctim,
                                    __bf16* __restrict__ w1t, __bf16* __restrict__ w2t) {
  int idx = blockIdx.x * blockDim.x + threadIdx.x;
  if (idx >= kL * kN * kH) return;
  int l = idx / (kN * kH);
  int r = idx % (kN * kH);
  int n = r / kH;   // output column
  int k = r % kH;   // reduction index
  float g = __expf(log_gamma[l * kN + n]);
  gBre[idx] = f2bf(g * B_re[idx]);         // Bn[n][k] * gamma[n]  (already [n][k])
  gBim[idx] = f2bf(g * B_im[idx]);
  Ctre[idx] = f2bf(C_re[idx]);             // C[h][n] already [col][k]
  Ctim[idx] = f2bf(C_im[idx]);
  w1t[idx] = f2bf(w1[((size_t)l * kH + k) * kH + n]);  // transpose w[k][j] -> [j][k]
  w2t[idx] = f2bf(w2[((size_t)l * kH + k) * kH + n]);
}

// ---------------- head: h[:, -1, :] @ out_w + out_b ----------------
__global__ void head_kernel(const float* __restrict__ h, const float* __restrict__ ow,
                            const float* __restrict__ ob, float* __restrict__ out) {
  int b = blockIdx.x, o = threadIdx.x;
  const float* hr = h + ((size_t)b * kT + (kT - 1)) * kH;
  float acc = ob[o];
  for (int k = 0; k < kH; ++k) acc += hr[k] * ow[k * kOUT + o];
  out[b * kOUT + o] = acc;
}

// ---------------- orchestration ----------------
extern "C" void kernel_launch(void* const* d_in, const int* in_sizes, int n_in,
                              void* d_out, int out_size, void* d_ws, size_t ws_size,
                              hipStream_t stream) {
  (void)in_sizes; (void)n_in; (void)out_size; (void)ws_size;
  const float* x         = (const float*)d_in[0];
  const float* emb_w     = (const float*)d_in[1];
  const float* emb_b     = (const float*)d_in[2];
  const float* log_nu    = (const float*)d_in[3];
  const float* log_theta = (const float*)d_in[4];
  const float* log_gamma = (const float*)d_in[5];
  const float* B_re      = (const float*)d_in[6];
  const float* B_im      = (const float*)d_in[7];
  const float* C_re      = (const float*)d_in[8];
  const float* C_im      = (const float*)d_in[9];
  const float* Dv        = (const float*)d_in[10];
  const float* w1        = (const float*)d_in[11];
  const float* b1        = (const float*)d_in[12];
  const float* w2        = (const float*)d_in[13];
  const float* b2        = (const float*)d_in[14];
  const float* out_w     = (const float*)d_in[15];
  const float* out_b     = (const float*)d_in[16];
  float* out = (float*)d_out;

  char* ws = (char*)d_ws;
  size_t off = 0;
  auto alloc = [&](size_t bytes) -> char* {
    char* p = ws + off;
    off += (bytes + 255) & ~(size_t)255;
    return p;
  };
  float*  h_f32 = (float*) alloc((size_t)kMT * kH * 4);
  __bf16* h_bf  = (__bf16*)alloc((size_t)kMT * kH * 2);
  float*  uRe   = (float*) alloc((size_t)kMT * kN * 4);
  float*  uIm   = (float*) alloc((size_t)kMT * kN * 4);
  __bf16* stRe  = (__bf16*)alloc((size_t)kMT * kN * 2);
  __bf16* stIm  = (__bf16*)alloc((size_t)kMT * kN * 2);
  __bf16* g_bf  = (__bf16*)alloc((size_t)kMT * kH * 2);
  float4* carry = (float4*)alloc((size_t)kB * kNCH * kN * 16);
  float2* pref  = (float2*)alloc((size_t)kB * kNCH * kN * 8);
  const size_t WSZ = (size_t)kL * kN * kH;
  __bf16* gBre = (__bf16*)alloc(WSZ * 2);
  __bf16* gBim = (__bf16*)alloc(WSZ * 2);
  __bf16* Ctre = (__bf16*)alloc(WSZ * 2);
  __bf16* Ctim = (__bf16*)alloc(WSZ * 2);
  __bf16* w1t  = (__bf16*)alloc(WSZ * 2);
  __bf16* w2t  = (__bf16*)alloc(WSZ * 2);

  prep_weights_kernel<<<(kL * kN * kH) / 256, 256, 0, stream>>>(
      log_gamma, B_re, B_im, C_re, C_im, w1, w2, gBre, gBim, Ctre, Ctim, w1t, w2t);
  embed_kernel<<<kMT, kH, 0, stream>>>(x, emb_w, emb_b, h_f32, h_bf);

  const dim3 gemm_grid(kH / BN, kMT / BM);
  for (int l = 0; l < kL; ++l) {
    const __bf16* gBre_l = gBre + (size_t)l * kN * kH;
    const __bf16* gBim_l = gBim + (size_t)l * kN * kH;
    const __bf16* Ctre_l = Ctre + (size_t)l * kH * kN;
    const __bf16* Ctim_l = Ctim + (size_t)l * kH * kN;
    const __bf16* w1t_l  = w1t  + (size_t)l * kH * kH;
    const __bf16* w2t_l  = w2t  + (size_t)l * kH * kH;

    // uB = h @ (gamma*B)^T  (re, im simultaneously)
    gemm_dual<<<gemm_grid, 256, 0, stream>>>(
        h_bf, nullptr, gBre_l, gBim_l, kH, kN, 0,
        uRe, uIm, nullptr, nullptr, nullptr, nullptr, nullptr, nullptr);

    // parallel complex scan over T
    scan_local_kernel<<<(kB * kNCH * kN) / 256, 256, 0, stream>>>(
        uRe, uIm, log_nu + l * kN, log_theta + l * kN, carry);
    scan_carry_kernel<<<(kB * kN) / 256, 256, 0, stream>>>(carry, pref);
    scan_apply_kernel<<<(kB * kNCH * kN) / 256, 256, 0, stream>>>(
        uRe, uIm, log_nu + l * kN, log_theta + l * kN, pref, stRe, stIm);

    // y = Re(states @ C^T) + h*D ; gelu -> g
    gemm_dual<<<gemm_grid, 256, 0, stream>>>(
        stRe, stIm, Ctre_l, Ctim_l, kN, kH, 1,
        nullptr, nullptr, h_f32, Dv + l * kH, g_bf, nullptr, nullptr, nullptr);

    // h = (g@w1+b1)*sigmoid(g@w2+b2) + h
    gemm_dual<<<gemm_grid, 256, 0, stream>>>(
        g_bf, nullptr, w1t_l, w2t_l, kH, kH, 2,
        nullptr, nullptr, nullptr, nullptr, h_bf, b1 + l * kH, b2 + l * kH, h_f32);
  }

  head_kernel<<<kB, kOUT, 0, stream>>>(h_f32, out_w, out_b, out);
}
